// SpatiallyVaryingPhysicalLayer_44976897524150
// MI455X (gfx1250) — matrix operand order repaired and checked by reading, not compiled, not run on hardware
//
#include <hip/hip_runtime.h>
#include <hip/hip_bf16.h>

typedef __attribute__((ext_vector_type(16))) _Float16 v16h;
typedef __attribute__((ext_vector_type(8)))  _Float16 v8h;
typedef __attribute__((ext_vector_type(8)))  float    v8f;

#define IMG_H 768
#define IMG_W 768
#define NBATCH 4
#define NCH 3
#define PSZ 128
#define STR 64
#define KSZ 31
#define NHP 11
#define NWP 11
#define NPATCH 121            // NHP*NWP
#define BN 484                // NBATCH*NPATCH
#define NPC 1452              // BN*NCH
#define NZ 14
#define HID 64
#define PADH 158              // PSZ + 2*15
#define PADW 176              // x0max(112)+Kmax(63)+1 -> 176, keeps 16B alignment
#define KROWP 80              // padded kernel row: index 15+d, d in [0,30], zeros elsewhere
#define TOTELEM (NBATCH*NCH*IMG_H*IMG_W)
#define EPSF 1e-6f
#define PI_F 3.14159265358979323846f

// ---------------------------------------------------------------- zero output
__global__ void zero_out_kernel(float* __restrict__ out) {
    int i = blockIdx.x * blockDim.x + threadIdx.x;
    if (i < TOTELEM) out[i] = 0.f;
}

// ------------------------------------------------- Zernike basis (run once, 1 block x 961)
__global__ void basis_kernel(float* __restrict__ basis) {
    __shared__ float s_val[NZ * 961];
    __shared__ float s_cnt;
    __shared__ float s_inv[NZ];
    const int tid = threadIdx.x;                 // 0..960
    const int si = tid / KSZ, sj = tid % KSZ;
    const float cy = -1.f + (float)si * (2.f / 30.f);
    const float cx = -1.f + (float)sj * (2.f / 30.f);
    const float m = (cx * cx + cy * cy <= 1.f) ? 1.f : 0.f;
    float px[5], py[5];
    px[0] = 1.f; py[0] = 1.f;
#pragma unroll
    for (int k = 1; k < 5; ++k) { px[k] = px[k - 1] * cx; py[k] = py[k - 1] * cy; }
    if (tid == 0) s_cnt = 0.f;
    __syncthreads();
    atomicAdd(&s_cnt, m);
    int z = 0;
#pragma unroll
    for (int deg = 1; deg <= 4; ++deg)
#pragma unroll
        for (int i2 = 0; i2 <= deg; ++i2) {
            s_val[z * 961 + tid] = px[deg - i2] * py[i2] * m;
            ++z;
        }
    __syncthreads();
    if (tid < NZ) {
        float s = 0.f;
        for (int q = 0; q < 961; ++q) { float v = s_val[tid * 961 + q]; s += v * v; }
        s_inv[tid] = 1.f / (sqrtf(s / s_cnt) + 1e-8f);
    }
    __syncthreads();
    for (int zz = 0; zz < NZ; ++zz) basis[zz * 961 + tid] = s_val[zz * 961 + tid] * s_inv[zz];
}

// ------------------------------------------------------------- per-patch MLP
__global__ __launch_bounds__(256) void mlp_kernel(const float* __restrict__ w1, const float* __restrict__ b1,
                                                  const float* __restrict__ w2, const float* __restrict__ b2,
                                                  const float* __restrict__ w3, const float* __restrict__ b3,
                                                  float* __restrict__ coeffs) {
    const int n = blockIdx.x * blockDim.x + threadIdx.x;
    if (n >= BN) return;
    const int t = n % NPATCH;
    const int nh = t / NWP, nw = t % NWP;
    const float cy = ((float)(nh * STR) + (float)PSZ * 0.5f) / (float)IMG_H * 2.f - 1.f;
    const float cx = ((float)(nw * STR) + (float)PSZ * 0.5f) / (float)IMG_W * 2.f - 1.f;
    float h1[HID], h2[HID];
    for (int j = 0; j < HID; ++j) h1[j] = tanhf(cy * w1[j] + cx * w1[HID + j] + b1[j]);
    for (int j = 0; j < HID; ++j) {
        float s = b2[j];
        for (int i = 0; i < HID; ++i) s += h1[i] * w2[i * HID + j];
        h2[j] = tanhf(s);
    }
    for (int z = 0; z < NZ; ++z) {
        float s = b3[z];
        for (int i = 0; i < HID; ++i) s += h2[i] * w3[i * NZ + z];
        coeffs[n * NZ + z] = s;
    }
}

// ----------------------------------------- PSF per (patch, wavelength): 31x31 DFT^2
__global__ __launch_bounds__(256) void psf_kernel(const float* __restrict__ coeffs,
                                                  const float* __restrict__ basis,
                                                  float* __restrict__ kern) {
    __shared__ float s_fr[961], s_fi[961], s_tr[961], s_ti[961];
    __shared__ float s_twr[KSZ], s_twi[KSZ];
    __shared__ float s_coef[NZ];
    __shared__ float s_sum;
    const int pc = blockIdx.x;
    const int n = pc / NCH, c = pc % NCH;
    const float wl = (c == 0) ? (0.53f / 0.61f) : ((c == 1) ? 1.0f : (0.53f / 0.47f));
    const int tid = threadIdx.x;
    if (tid < NZ) s_coef[tid] = coeffs[n * NZ + tid];
    if (tid < KSZ) {
        float ang = -2.f * PI_F * (float)tid / (float)KSZ;   // forward DFT: e^{-2*pi*i*k/31}
        __sincosf(ang, &s_twi[tid], &s_twr[tid]);
    }
    if (tid == 0) s_sum = 0.f;
    __syncthreads();
    // pupil field, ifftshift folded into index map: work[k] = field[(k+15)%31]
    for (int idx = tid; idx < 961; idx += 256) {
        int k1 = idx / KSZ, k2 = idx % KSZ;
        int si = (k1 + 15) % KSZ, sj = (k2 + 15) % KSZ;
        float cy = -1.f + (float)si * (2.f / 30.f);
        float cx = -1.f + (float)sj * (2.f / 30.f);
        float m = (cx * cx + cy * cy <= 1.f) ? 1.f : 0.f;
        float ph = 0.f;
        for (int z = 0; z < NZ; ++z) ph += s_coef[z] * basis[z * 961 + si * KSZ + sj];
        float sn, cs;
        __sincosf(2.f * PI_F * wl * ph, &sn, &cs);
        s_fr[idx] = m * cs;
        s_fi[idx] = m * sn;
    }
    __syncthreads();
    // row DFT
    for (int idx = tid; idx < 961; idx += 256) {
        int i = idx / KSZ, f = idx % KSZ;
        float ar = 0.f, ai = 0.f;
        int p = 0;
        for (int j = 0; j < KSZ; ++j) {
            float wr = s_twr[p], wi = s_twi[p];
            float xr = s_fr[i * KSZ + j], xi = s_fi[i * KSZ + j];
            ar += xr * wr - xi * wi;
            ai += xr * wi + xi * wr;
            p += f; if (p >= KSZ) p -= KSZ;
        }
        s_tr[idx] = ar; s_ti[idx] = ai;
    }
    __syncthreads();
    // column DFT + |.|^2 with fftshift folded: psf[(f+15)%31]
    for (int idx = tid; idx < 961; idx += 256) {
        int f1 = idx / KSZ, f2 = idx % KSZ;
        float ar = 0.f, ai = 0.f;
        int p = 0;
        for (int k1 = 0; k1 < KSZ; ++k1) {
            float wr = s_twr[p], wi = s_twi[p];
            float xr = s_tr[k1 * KSZ + f2], xi = s_ti[k1 * KSZ + f2];
            ar += xr * wr - xi * wi;
            ai += xr * wi + xi * wr;
            p += f1; if (p >= KSZ) p -= KSZ;
        }
        float pw = ar * ar + ai * ai;
        s_fr[((f1 + 15) % KSZ) * KSZ + (f2 + 15) % KSZ] = pw;   // s_fr reused as psf
        atomicAdd(&s_sum, pw);
    }
    __syncthreads();
    const float inv = 1.f / (s_sum + EPSF);
    float* kd = kern + (long)pc * (KSZ * KSZ);
    // normalize + flip both axes (reference does kernels[..., ::-1, ::-1])
    for (int idx = tid; idx < 961; idx += 256) {
        int i = idx / KSZ, j = idx % KSZ;
        kd[idx] = s_fr[(30 - i) * KSZ + (30 - j)] * inv;
    }
}

// ------------------- WMMA depthwise conv: one block per (patch, channel), 8 waves
__global__ __launch_bounds__(256) void conv_kernel(const float* __restrict__ x,
                                                   const float* __restrict__ kern,
                                                   const float* __restrict__ gptr,
                                                   float* __restrict__ out) {
    __shared__ _Float16 s_patch[PADH * PADW];    // zero-padded patch, f16 (55.6 KB)
    __shared__ _Float16 s_krow[KSZ * KROWP];     // zero-padded kernel rows
    __shared__ float    s_hann[PSZ];
    const int pc = blockIdx.x;
    const int n = pc / NCH, c = pc % NCH;
    const int b = n / NPATCH, t = n % NPATCH;
    const int nh = t / NWP, nw = t % NWP;
    const int tid = threadIdx.x;
    const float gamma = gptr[0];

    for (int i = tid; i < PADH * PADW / 2; i += 256) ((unsigned int*)s_patch)[i] = 0u;
    for (int i = tid; i < KSZ * KROWP / 2; i += 256) ((unsigned int*)s_krow)[i] = 0u;
    if (tid < PSZ) s_hann[tid] = 0.5f * (1.f - __cosf(2.f * PI_F * (float)tid / (float)PSZ));
    __syncthreads();

    // gamma-encode patch into padded LDS (f16)
    const float* src = x + (((long)(b * NCH + c)) * IMG_H + nh * STR) * IMG_W + nw * STR;
    for (int i = tid; i < PSZ * PSZ; i += 256) {
        int py = i >> 7, pxc = i & (PSZ - 1);
        float v = src[py * IMG_W + pxc];
        v = powf(fmaxf(v, 0.f) + EPSF, gamma);
        s_patch[(py + 15) * PADW + (pxc + 15)] = (_Float16)v;
    }
    const float* kr = kern + (long)pc * (KSZ * KSZ);
    for (int i = tid; i < KSZ * KSZ; i += 256) {
        int ky = i / KSZ, kx = i % KSZ;
        s_krow[ky * KROWP + 15 + kx] = (_Float16)kr[i];
    }
    __syncthreads();

    const int lane = tid & 31;
    const int wv   = tid >> 5;       // wave = tile row (8 waves x 8 tiles = 64 tiles)
    const int nidx = lane & 15;      // A: M row / B: N col / D: N col
    const int grp  = lane >> 4;
    const int y0   = wv * 16;

    v8f acc[8];
#pragma unroll
    for (int q = 0; q < 8; ++q)
#pragma unroll
        for (int e = 0; e < 8; ++e) acc[q][e] = 0.f;

    for (int ky = 0; ky < KSZ; ++ky) {
        // B = Toeplitz band of kernel row ky: T[K,N] = krow[K-N]; element e -> K = (e&8)*2 + 8*grp + (e&7) (+32)
        v16h bf0, bf1;
        const int kb = ky * KROWP + 15 + 8 * grp - nidx;
#pragma unroll
        for (int e = 0; e < 16; ++e) { bf0[e] = s_krow[kb + e + (e & 8)]; bf1[e] = s_krow[kb + 32 + e + (e & 8)]; }
        const _Float16* prow = &s_patch[(y0 + nidx + ky) * PADW];
#pragma unroll
        for (int tx = 0; tx < 8; ++tx) {
            const int x0 = tx * 16;
            // A element e -> K = (e&8)*2 + 8*grp + (e&7): two contiguous 8-half runs (16B aligned)
            v8h lo0 = *(const v8h*)(prow + x0 +  0 + 8 * grp);
            v8h hi0 = *(const v8h*)(prow + x0 + 16 + 8 * grp);
            v8h lo1 = *(const v8h*)(prow + x0 + 32 + 8 * grp);
            v8h hi1 = *(const v8h*)(prow + x0 + 48 + 8 * grp);
            v16h af0, af1;
#pragma unroll
            for (int e = 0; e < 8; ++e) {
                af0[e] = lo0[e]; af0[8 + e] = hi0[e];
                af1[e] = lo1[e]; af1[8 + e] = hi1[e];
            }
            acc[tx] = __builtin_amdgcn_wmma_f32_16x16x32_f16(false, af0, false, bf0,
                                                             (short)0, acc[tx], false, false);
            acc[tx] = __builtin_amdgcn_wmma_f32_16x16x32_f16(false, af1, false, bf1,
                                                             (short)0, acc[tx], false, false);
        }
    }

    // Hann window + overlap-add (f32 atomics into d_out)
    float* dst = out + ((long)(b * NCH + c)) * IMG_H * IMG_W;
#pragma unroll
    for (int tx = 0; tx < 8; ++tx) {
        const int xl = tx * 16 + nidx;
        const int gx = nw * STR + xl;
        const float wx = s_hann[xl];
#pragma unroll
        for (int i = 0; i < 8; ++i) {
            const int yl = y0 + i + 8 * grp;     // D layout: VGPR i -> M = i + 8*grp
            const float v = acc[tx][i] * wx * s_hann[yl];
            atomicAdd(&dst[(nh * STR + yl) * IMG_W + gx], v);
        }
    }
}

// --------------------------------------------- normalize + gamma-decode in place
__device__ __forceinline__ float hannf(int p) {
    return 0.5f * (1.f - __cosf(2.f * PI_F * (float)p / (float)PSZ));
}

__global__ void finalize_kernel(const float* __restrict__ gptr, float* __restrict__ out) {
    int i = blockIdx.x * blockDim.x + threadIdx.x;
    if (i >= TOTELEM) return;
    const int gx = i % IMG_W;
    const int gy = (i / IMG_W) % IMG_H;
    const float invg = 1.f / gptr[0];
    int nhmin = (gy >= PSZ) ? (gy - (PSZ - 1) + (STR - 1)) / STR : 0;
    int nhmax = min(NHP - 1, gy / STR);
    float sy = 0.f;
    for (int nh = nhmin; nh <= nhmax; ++nh) sy += hannf(gy - nh * STR);
    int nwmin = (gx >= PSZ) ? (gx - (PSZ - 1) + (STR - 1)) / STR : 0;
    int nwmax = min(NWP - 1, gx / STR);
    float sx = 0.f;
    for (int nw = nwmin; nw <= nwmax; ++nw) sx += hannf(gx - nw * STR);
    float v = out[i] / (sy * sx + EPSF);
    out[i] = powf(fmaxf(v, 0.f) + EPSF, invg);
}

extern "C" void kernel_launch(void* const* d_in, const int* in_sizes, int n_in,
                              void* d_out, int out_size, void* d_ws, size_t ws_size,
                              hipStream_t stream) {
    (void)in_sizes; (void)n_in; (void)out_size; (void)ws_size;
    const float* x  = (const float*)d_in[0];
    const float* w1 = (const float*)d_in[1];
    const float* b1 = (const float*)d_in[2];
    const float* w2 = (const float*)d_in[3];
    const float* b2 = (const float*)d_in[4];
    const float* w3 = (const float*)d_in[5];
    const float* b3 = (const float*)d_in[6];
    const float* g  = (const float*)d_in[7];
    float* out = (float*)d_out;

    float* wsf    = (float*)d_ws;
    float* basis  = wsf;                  // 14*961 floats
    float* coeffs = basis + NZ * 961;     // 484*14 floats
    float* kern   = coeffs + BN * NZ;     // 1452*961 floats (~5.6 MB total ws)

    zero_out_kernel<<<dim3((TOTELEM + 255) / 256), dim3(256), 0, stream>>>(out);
    basis_kernel<<<dim3(1), dim3(961), 0, stream>>>(basis);
    mlp_kernel<<<dim3((BN + 255) / 256), dim3(256), 0, stream>>>(w1, b1, w2, b2, w3, b3, coeffs);
    psf_kernel<<<dim3(NPC), dim3(256), 0, stream>>>(coeffs, basis, kern);
    conv_kernel<<<dim3(NPC), dim3(256), 0, stream>>>(x, kern, g, out);
    finalize_kernel<<<dim3((TOTELEM + 255) / 256), dim3(256), 0, stream>>>(g, out);
}
